// ExplicitMemoryBlock_73521250173368
// MI455X (gfx1250) — matrix operand order repaired and checked by reading, not compiled, not run on hardware
//
#include <hip/hip_runtime.h>
#include <hip/hip_bf16.h>
#include <math.h>

typedef __attribute__((ext_vector_type(16))) _Float16 v16h;
typedef __attribute__((ext_vector_type(8)))  _Float16 v8h;
typedef __attribute__((ext_vector_type(8)))  float    v8f;

union HF { v16h v; v8h h[2]; };

// ---------------------------------------------------------------------------
// WMMA helpers (CDNA5 gfx1250, wave32, V_WMMA_F32_16X16X32_F16)
// ---------------------------------------------------------------------------
__device__ __forceinline__ v8f wmma16(v16h a, v16h b, v8f c) {
  return __builtin_amdgcn_wmma_f32_16x16x32_f16(
      /*neg_a=*/false, a, /*neg_b=*/false, b,
      /*c_mod=*/(short)0, c, /*reuse_a=*/false, /*reuse_b=*/false);
}

// A fragment (16x32 f16) from an LDS tile that is K-contiguous per row.
// lane<16: row=lane, K={kb..kb+7, kb+16..kb+23}, kb=0; lane>=16: kb=8.
// Two 16-byte contiguous runs -> 2x ds_load_b128.
__device__ __forceinline__ v16h frag_a(const _Float16* p, int ld, int lane) {
  const int r  = lane & 15;
  const int kb = (lane & 16) ? 8 : 0;
  const _Float16* q = p + r * ld + kb;
  HF f;
  f.h[0] = *(const v8h*)(q);
  f.h[1] = *(const v8h*)(q + 16);
  return f.v;
}

// B fragment (32x16, K x N) from a tile stored TRANSPOSED in LDS:
// element (k, n) lives at p[n*ld + k].  lane n'=lane&15 needs K={kb..kb+7,
// kb+16..kb+23} (kb = 0 or 8) at fixed n' -> 2x ds_load_b128.
__device__ __forceinline__ v16h frag_bt(const _Float16* p, int ld, int lane) {
  const int n  = lane & 15;
  const int kb = (lane & 16) ? 8 : 0;
  const _Float16* q = p + n * ld + kb;
  HF f;
  f.h[0] = *(const v8h*)(q);
  f.h[1] = *(const v8h*)(q + 16);
  return f.v;
}

// ---------------------------------------------------------------------------
// Generic tiled WMMA GEMM (software-pipelined double-buffered LDS):
// out = act(A[f16] @ W + bias) (+Cadd)
// A : Mrows x Kdim (row-major);  Wt : N x Kdim (row-major == W transposed).
// Mrows % 128 == 0, N % 128 == 0, Kdim % 32 == 0.
// act: 0=none, 1=exact gelu, 2=sigmoid
// ---------------------------------------------------------------------------
__global__ __launch_bounds__(256) void gemm_wmma(
    const _Float16* __restrict__ A, const _Float16* __restrict__ Wt,
    const float* __restrict__ bias, const float* __restrict__ Cadd,
    float* __restrict__ out32, _Float16* __restrict__ out16,
    int Mrows, int Kdim, int N, int act)
{
  __shared__ __align__(16) _Float16 As[2][128][32];   // M x K slices
  __shared__ __align__(16) _Float16 Bts[2][128][32];  // N x K slices (W^T)

  const int tid  = threadIdx.x;
  const int lane = tid & 31;
  const int wid  = tid >> 5;
  const long bm  = (long)blockIdx.x * 128;
  const long bn  = (long)blockIdx.y * 128;
  const int  wm  = (wid & 3) * 32;   // 4 wave-rows of 32
  const int  wn  = (wid >> 2) * 64;  // 2 wave-cols of 64

  // per-thread staging coordinates: 2 x (128 rows x 4 uint4)
  const int r0 = tid >> 2,            c0 = (tid & 3) * 8;
  const int r1 = (tid + 256) >> 2,    c1 = ((tid + 256) & 3) * 8;
  const _Float16* gA0 = &A[(bm + r0) * (long)Kdim + c0];
  const _Float16* gB0 = &Wt[(bn + r0) * (long)Kdim + c0];
  const _Float16* gA1 = &A[(bm + r1) * (long)Kdim + c1];
  const _Float16* gB1 = &Wt[(bn + r1) * (long)Kdim + c1];

  v8f zero = {};
  v8f acc[2][4];
#pragma unroll
  for (int mi = 0; mi < 2; ++mi)
#pragma unroll
    for (int ni = 0; ni < 4; ++ni) acc[mi][ni] = zero;

  // prologue: stage k0 = 0 into buffer 0
  *(uint4*)(&As[0][r0][c0])  = *(const uint4*)(gA0);
  *(uint4*)(&Bts[0][r0][c0]) = *(const uint4*)(gB0);
  *(uint4*)(&As[0][r1][c1])  = *(const uint4*)(gA1);
  *(uint4*)(&Bts[0][r1][c1]) = *(const uint4*)(gB1);
  __syncthreads();

  int cur = 0;
  for (int k0 = 0; k0 < Kdim; k0 += 32, cur ^= 1) {
    const int nxt = cur ^ 1;
    const bool have_next = (k0 + 32) < Kdim;
    const int  kn = k0 + 32;

    // 1) issue next-tile global loads into distinct registers (stay in flight)
    uint4 na0, nb0, na1, nb1;
    if (have_next) {
      na0 = *(const uint4*)(gA0 + kn);
      nb0 = *(const uint4*)(gB0 + kn);
      na1 = *(const uint4*)(gA1 + kn);
      nb1 = *(const uint4*)(gB1 + kn);
      if (k0 + 64 < Kdim) {                           // global_prefetch hints
        __builtin_prefetch(gA0 + k0 + 64, 0, 3);
        __builtin_prefetch(gB0 + k0 + 64, 0, 3);
      }
    }

    // 2) consume current buffer with WMMAs (hides the global load latency)
    v16h af[2], bf[4];
    af[0] = frag_a(&As[cur][wm][0],      32, lane);
    af[1] = frag_a(&As[cur][wm + 16][0], 32, lane);
#pragma unroll
    for (int ni = 0; ni < 4; ++ni)
      bf[ni] = frag_bt(&Bts[cur][wn + ni * 16][0], 32, lane);
#pragma unroll
    for (int mi = 0; mi < 2; ++mi)
#pragma unroll
      for (int ni = 0; ni < 4; ++ni)
        acc[mi][ni] = wmma16(af[mi], bf[ni], acc[mi][ni]);

    // 3) commit next tile to LDS, then one barrier per iteration
    if (have_next) {
      *(uint4*)(&As[nxt][r0][c0])  = na0;
      *(uint4*)(&Bts[nxt][r0][c0]) = nb0;
      *(uint4*)(&As[nxt][r1][c1])  = na1;
      *(uint4*)(&Bts[nxt][r1][c1]) = nb1;
    }
    __syncthreads();  // nxt stores visible; cur reads done before reuse
  }

  const int rlo = (lane & 16) ? 8 : 0;
  const int cl  = lane & 15;
#pragma unroll
  for (int mi = 0; mi < 2; ++mi)
#pragma unroll
    for (int ni = 0; ni < 4; ++ni)
#pragma unroll
      for (int r = 0; r < 8; ++r) {
        long row = bm + wm + mi * 16 + r + rlo;
        long col = bn + wn + ni * 16 + cl;
        float v = acc[mi][ni][r];
        if (bias) v += bias[col];
        if (act == 1)      v = 0.5f * v * (1.0f + erff(v * 0.70710678118f));
        else if (act == 2) v = 1.0f / (1.0f + expf(-v));
        if (Cadd)  v += Cadd[row * N + col];
        if (out32) out32[row * N + col] = v;
        if (out16) out16[row * N + col] = (_Float16)v;
      }
}

// ---------------------------------------------------------------------------
// LayerNorm: one block per token row
// ---------------------------------------------------------------------------
__global__ __launch_bounds__(256) void ln_kernel(
    const float* __restrict__ x, const float* __restrict__ g,
    const float* __restrict__ bta, _Float16* __restrict__ xn16, int Dd)
{
  __shared__ float red[256];
  const long row = blockIdx.x;
  const float* xr = x + row * Dd;
  const int tid = threadIdx.x;
  float s = 0.f, s2 = 0.f;
  for (int c = tid; c < Dd; c += 256) { float v = xr[c]; s += v; s2 += v * v; }
  red[tid] = s; __syncthreads();
  for (int o = 128; o > 0; o >>= 1) { if (tid < o) red[tid] += red[tid + o]; __syncthreads(); }
  float mean = red[0] / (float)Dd; __syncthreads();
  red[tid] = s2; __syncthreads();
  for (int o = 128; o > 0; o >>= 1) { if (tid < o) red[tid] += red[tid + o]; __syncthreads(); }
  float var = red[0] / (float)Dd - mean * mean;
  float inv = rsqrtf(var + 1e-5f);
  for (int c = tid; c < Dd; c += 256)
    xn16[row * Dd + c] = (_Float16)((xr[c] - mean) * inv * g[c] + bta[c]);
}

// ---------------------------------------------------------------------------
// write_importance = sigmoid(xn @ wg_w + wg_b); one block per row
// ---------------------------------------------------------------------------
__global__ __launch_bounds__(256) void imp_kernel(
    const _Float16* __restrict__ xn16, const float* __restrict__ wg_w,
    const float* __restrict__ wg_b, float* __restrict__ imp, int Dd)
{
  __shared__ float red[256];
  const long row = blockIdx.x;
  const int tid = threadIdx.x;
  float s = 0.f;
  for (int c = tid; c < Dd; c += 256) s += (float)xn16[row * Dd + c] * wg_w[c];
  red[tid] = s; __syncthreads();
  for (int o = 128; o > 0; o >>= 1) { if (tid < o) red[tid] += red[tid + o]; __syncthreads(); }
  if (tid == 0) imp[row] = 1.0f / (1.0f + expf(-(red[0] + wg_b[0])));
}

// ---------------------------------------------------------------------------
// Per-batch bitonic top-k (descending, tie-break on index) in LDS
// ---------------------------------------------------------------------------
__global__ __launch_bounds__(1024) void topk_kernel(
    const float* __restrict__ imp, int* __restrict__ topidx, int S, int NW)
{
  __shared__ float sv[4096];
  __shared__ int   si[4096];
  const int b = blockIdx.x, tid = threadIdx.x;
  for (int i = tid; i < S; i += 1024) { sv[i] = imp[(long)b * S + i]; si[i] = i; }
  __syncthreads();
  for (int k = 2; k <= S; k <<= 1)
    for (int j = k >> 1; j > 0; j >>= 1) {
      for (int i = tid; i < S; i += 1024) {
        int l = i ^ j;
        if (l > i) {
          bool desc = ((i & k) == 0);
          float a = sv[i], c = sv[l];
          bool sw = desc ? (a < c) : (a > c);
          if (a == c) sw = desc ? (si[i] > si[l]) : (si[i] < si[l]);
          if (sw) { sv[i] = c; sv[l] = a; int t = si[i]; si[i] = si[l]; si[l] = t; }
        }
      }
      __syncthreads();
    }
  for (int i = tid; i < NW; i += 1024) topidx[b * NW + i] = si[i];
}

// ---------------------------------------------------------------------------
// Build updated_memory (f32 output) + f16 stored keys (row-major) and
// stored values TRANSPOSED (B, V, M) so flash P@V tiles stage contiguously.
// ---------------------------------------------------------------------------
__global__ void build_mem_kernel(
    const float* __restrict__ mem, const float* __restrict__ mk32,
    const float* __restrict__ mv32, const int* __restrict__ topidx,
    float* __restrict__ um, _Float16* __restrict__ sk16, _Float16* __restrict__ svT16,
    int Bb, int Mm, int KV, int Kd, int keep, int S, int NW)
{
  long idx = (long)blockIdx.x * 256 + threadIdx.x;
  long total = (long)Bb * Mm * KV;
  if (idx >= total) return;
  int c = (int)(idx % KV);
  long t = idx / KV;
  int m = (int)(t % Mm);
  int b = (int)(t / Mm);
  float val;
  if (m < keep) val = mem[idx];
  else {
    int s = topidx[b * NW + (m - keep)];
    val = (c < Kd) ? mk32[((long)b * S + s) * Kd + c]
                   : mv32[((long)b * S + s) * Kd + (c - Kd)];
  }
  um[idx] = val;
  if (c < Kd) sk16[((long)b * Mm + m) * Kd + c] = (_Float16)val;
  else        svT16[((long)b * (KV - Kd) + (c - Kd)) * Mm + m] = (_Float16)val;
}

// ---------------------------------------------------------------------------
// Flash attention over memory: 64 q-rows / block (4 waves, 16 rows each),
// 64-key chunks. QK^T and P@V via WMMA; all fragments 128-bit LDS loads.
// ---------------------------------------------------------------------------
__global__ __launch_bounds__(128) void flash_attn_kernel(
    const _Float16* __restrict__ Qm,  // (B,S,128)
    const _Float16* __restrict__ Km,  // (B,M,128)
    const _Float16* __restrict__ VT,  // (B,128,M)  values transposed
    _Float16* __restrict__ Om,        // (B,S,128)
    int S, int Mtot)
{
  __shared__ __align__(16) _Float16 Qs[64][128];   // (qrow, kdim)
  __shared__ __align__(16) _Float16 Ks[64][128];   // (key, kdim)
  __shared__ __align__(16) _Float16 Vst[128][64];  // (vdim, key)
  __shared__ __align__(16) _Float16 Ps[4][16][64]; // (wave, qrow, key)

  const int b    = blockIdx.y;
  const int row0 = blockIdx.x * 64;
  const int tid  = threadIdx.x, lane = tid & 31, wid = tid >> 5;
  const int qr   = wid * 16;
  const int rlo  = (lane & 16) ? 8 : 0;
  const int cl   = lane & 15;
  const float scale = 0.08838834764831845f; // 1/sqrt(128)

#pragma unroll
  for (int l = 0; l < 8; ++l) {
    int idx = tid + l * 128; int r = idx >> 4, c = (idx & 15) * 8;
    *(uint4*)&Qs[r][c] = *(const uint4*)&Qm[((long)b * S + row0 + r) * 128 + c];
  }

  v8f zero = {};
  float row_m[8], row_l[8];
  v8f Oacc[8];
#pragma unroll
  for (int r = 0; r < 8; ++r) { row_m[r] = -1e30f; row_l[r] = 0.f; }
#pragma unroll
  for (int vt = 0; vt < 8; ++vt) Oacc[vt] = zero;

  for (int m0 = 0; m0 < Mtot; m0 += 64) {
    __syncthreads();
#pragma unroll
    for (int l = 0; l < 8; ++l) {
      int idx = tid + l * 128;
      int rk = idx >> 4, ck = (idx & 15) * 8;      // Ks: 64 rows x 16 uint4
      *(uint4*)&Ks[rk][ck] =
          *(const uint4*)&Km[((long)b * Mtot + m0 + rk) * 128 + ck];
      int rv = idx >> 3, cv = (idx & 7) * 8;       // Vst: 128 rows x 8 uint4
      *(uint4*)&Vst[rv][cv] =
          *(const uint4*)&VT[((long)b * 128 + rv) * Mtot + m0 + cv];
    }
    __syncthreads();

    // S = Q @ K^T over 64 keys (4 tiles of 16), kdim 128 (4 wmma steps)
    v8f Sacc[4];
#pragma unroll
    for (int nt = 0; nt < 4; ++nt) Sacc[nt] = zero;
#pragma unroll
    for (int nt = 0; nt < 4; ++nt)
#pragma unroll
      for (int kk = 0; kk < 4; ++kk)
        Sacc[nt] = wmma16(frag_a(&Qs[qr][kk * 32], 128, lane),
                          frag_bt(&Ks[nt * 16][kk * 32], 128, lane), Sacc[nt]);

    // online softmax (each lane owns one row per C-register index r)
#pragma unroll
    for (int r = 0; r < 8; ++r) {
      float mx = -1e30f;
#pragma unroll
      for (int nt = 0; nt < 4; ++nt) mx = fmaxf(mx, Sacc[nt][r]);
      mx *= scale;
      for (int off = 1; off < 16; off <<= 1) mx = fmaxf(mx, __shfl_xor(mx, off, 32));
      float m_new = fmaxf(row_m[r], mx);
      float alpha = expf(row_m[r] - m_new);
      float ssum  = 0.f;
#pragma unroll
      for (int nt = 0; nt < 4; ++nt) {
        float p = expf(Sacc[nt][r] * scale - m_new);
        Sacc[nt][r] = p; ssum += p;
      }
      for (int off = 1; off < 16; off <<= 1) ssum += __shfl_xor(ssum, off, 32);
      row_l[r] = row_l[r] * alpha + ssum;
      row_m[r] = m_new;
#pragma unroll
      for (int vt = 0; vt < 8; ++vt) Oacc[vt][r] *= alpha;
    }

    // stage P into LDS so it can re-enter as an A-fragment
#pragma unroll
    for (int nt = 0; nt < 4; ++nt)
#pragma unroll
      for (int r = 0; r < 8; ++r)
        Ps[wid][r + rlo][nt * 16 + cl] = (_Float16)Sacc[nt][r];
    __syncthreads();

    // O += P(16x64) @ V(64x128); V tile is (vdim,key) -> contiguous B frags
#pragma unroll
    for (int vt = 0; vt < 8; ++vt)
#pragma unroll
      for (int kk = 0; kk < 2; ++kk)
        Oacc[vt] = wmma16(frag_a(&Ps[wid][0][kk * 32], 64, lane),
                          frag_bt(&Vst[vt * 16][kk * 32], 64, lane), Oacc[vt]);
  }

#pragma unroll
  for (int vt = 0; vt < 8; ++vt)
#pragma unroll
    for (int r = 0; r < 8; ++r) {
      long row = row0 + qr + r + rlo;
      Om[((long)b * S + row) * 128 + vt * 16 + cl] =
          (_Float16)(Oacc[vt][r] / row_l[r]);
    }
}

// ---------------------------------------------------------------------------
// Column mean of queries over S, then global-context projection
// ---------------------------------------------------------------------------
__global__ void colmean_kernel(const _Float16* __restrict__ q16,
                               float* __restrict__ qmean, int S, int Dd)
{
  int idx = blockIdx.x * 256 + threadIdx.x; // over B*D
  int b = idx / Dd, c = idx % Dd;
  float s = 0.f;
  for (int t = 0; t < S; ++t) s += (float)q16[((long)b * S + t) * Dd + c];
  qmean[idx] = s / (float)S;
}

__global__ void gp_kernel(const float* __restrict__ qmean,
                          const float* __restrict__ gp_w,
                          const float* __restrict__ gp_b,
                          float* __restrict__ gctx, int Dd)
{
  int idx = blockIdx.x * 256 + threadIdx.x; // over B*D
  int b = idx / Dd, n = idx % Dd;
  float s = gp_b[n];
  for (int c = 0; c < Dd; ++c) s += qmean[b * Dd + c] * gp_w[(long)c * Dd + n];
  gctx[idx] = s;
}

// ---------------------------------------------------------------------------
// Grouped conv1d (k=3, pad=1, groups=8) fused with combine:
// comb = (local + 0.1*gctx + 0.5*mem_out) * sigmoid(keys)
// ---------------------------------------------------------------------------
__global__ __launch_bounds__(256) void conv_combine_kernel(
    const _Float16* __restrict__ vv, const float* __restrict__ conv_w,
    const float* __restrict__ conv_b, const float* __restrict__ gctx,
    const _Float16* __restrict__ mo16, const _Float16* __restrict__ k16,
    _Float16* __restrict__ comb16, int S, int Dd, int Cg)
{
  __shared__ _Float16 vrow[3][1024];
  const int bs = blockIdx.x;
  const int b = bs / S, s = bs % S;
  const int tid = threadIdx.x;
  for (int t = 0; t < 3; ++t) {
    int ss = s + t - 1;
    for (int c = tid; c < Dd; c += 256)
      vrow[t][c] = (ss >= 0 && ss < S) ? vv[((long)b * S + ss) * Dd + c]
                                       : (_Float16)0.0f;
  }
  __syncthreads();
  for (int cc = tid; cc < Dd; cc += 256) {
    int g = cc / Cg;
    const float* wq = conv_w + (long)cc * Cg * 3;
    float acc = conv_b[cc];
    for (int ci = 0; ci < Cg; ++ci) {
      int cin = g * Cg + ci;
      acc += wq[ci * 3 + 0] * (float)vrow[0][cin]
           + wq[ci * 3 + 1] * (float)vrow[1][cin]
           + wq[ci * 3 + 2] * (float)vrow[2][cin];
    }
    long o = ((long)b * S + s) * Dd + cc;
    float comb = acc + 0.1f * gctx[b * Dd + cc] + 0.5f * (float)mo16[o];
    float pw = 1.0f / (1.0f + expf(-(float)k16[o]));
    comb16[o] = (_Float16)(comb * pw);
  }
}

// ---------------------------------------------------------------------------
// Weight convert+transpose: src (K x N) f32 -> dst (N x K) f16
// ---------------------------------------------------------------------------
__global__ void transpose_f32_to_f16(const float* __restrict__ src,
                                     _Float16* __restrict__ dst, int K, int N)
{
  long idx = (long)blockIdx.x * 256 + threadIdx.x;
  if (idx >= (long)K * N) return;
  int n = (int)(idx / K), k = (int)(idx % K);
  dst[idx] = (_Float16)src[(long)k * N + n];
}

// ---------------------------------------------------------------------------
// Launch
// ---------------------------------------------------------------------------
extern "C" void kernel_launch(void* const* d_in, const int* in_sizes, int n_in,
                              void* d_out, int out_size, void* d_ws, size_t ws_size,
                              hipStream_t stream) {
  (void)in_sizes; (void)n_in; (void)out_size; (void)ws_size;

  const int Bc = 4, Sc = 4096, Dc = 1024, Kc = 128, Vc = 128, Mc = 2048;
  const int NWc = 1024, KEEP = Mc - NWc, KVc = Kc + Vc, GR = 8;
  const long BSc = (long)Bc * Sc;

  const float* x       = (const float*)d_in[0];
  const float* mem     = (const float*)d_in[1];
  const float* ln_g    = (const float*)d_in[2];
  const float* ln_b    = (const float*)d_in[3];
  const float* mk_w1   = (const float*)d_in[4];
  const float* mk_b1   = (const float*)d_in[5];
  const float* mk_w2   = (const float*)d_in[6];
  const float* mk_b2   = (const float*)d_in[7];
  const float* mv_w1   = (const float*)d_in[8];
  const float* mv_b1   = (const float*)d_in[9];
  const float* mv_w2   = (const float*)d_in[10];
  const float* mv_b2   = (const float*)d_in[11];
  const float* mq_w1   = (const float*)d_in[12];
  const float* mq_b1   = (const float*)d_in[13];
  const float* mq_w2   = (const float*)d_in[14];
  const float* mq_b2   = (const float*)d_in[15];
  const float* wg_w    = (const float*)d_in[16];
  const float* wg_b    = (const float*)d_in[17];
  const float* q_w     = (const float*)d_in[18];
  const float* k_w     = (const float*)d_in[19];
  const float* v_w     = (const float*)d_in[20];
  const float* conv_w  = (const float*)d_in[21];
  const float* conv_b  = (const float*)d_in[22];
  const float* gp_w    = (const float*)d_in[23];
  const float* gp_b    = (const float*)d_in[24];
  const float* mr_w    = (const float*)d_in[25];
  const float* mr_b    = (const float*)d_in[26];
  const float* out_w   = (const float*)d_in[27];
  const float* out_b   = (const float*)d_in[28];

  float* out_main = (float*)d_out;                  // (B,S,D)
  float* out_mem  = out_main + BSc * Dc;            // (B,M,K+V)

  char* wsb = (char*)d_ws;
  size_t off = 0;
  auto alloc = [&](size_t bytes) -> void* {
    void* p = wsb + off;
    off += (bytes + 255) & ~(size_t)255;
    return p;
  };

  _Float16* xn16   = (_Float16*)alloc(BSc * Dc * 2);
  _Float16* tmp16  = (_Float16*)alloc(BSc * Dc * 2);
  _Float16* q16    = (_Float16*)alloc(BSc * Dc * 2);
  _Float16* k16    = (_Float16*)alloc(BSc * Dc * 2);
  _Float16* v16b   = (_Float16*)alloc(BSc * Dc * 2);
  _Float16* mo16   = (_Float16*)alloc(BSc * Dc * 2);
  _Float16* comb16 = (_Float16*)alloc(BSc * Dc * 2);
  _Float16* mq16   = (_Float16*)alloc(BSc * Kc * 2);
  _Float16* ret16  = (_Float16*)alloc(BSc * Vc * 2);
  float*    mk32   = (float*)alloc(BSc * Kc * 4);
  float*    mv32   = (float*)alloc(BSc * Kc * 4);
  _Float16* sk16   = (_Float16*)alloc((long)Bc * Mc * Kc * 2);
  _Float16* svT16  = (_Float16*)alloc((long)Bc * Vc * Mc * 2);
  float*    imp    = (float*)alloc(BSc * 4);
  int*      topidx = (int*)alloc((long)Bc * NWc * 4);
  float*    qmean  = (float*)alloc((long)Bc * Dc * 4);
  float*    gctx   = (float*)alloc((long)Bc * Dc * 4);
  // transposed f16 weights (N x K layout)
  _Float16* w_mk1 = (_Float16*)alloc((long)Dc * Dc * 2);
  _Float16* w_mk2 = (_Float16*)alloc((long)Kc * Dc * 2);
  _Float16* w_mv1 = (_Float16*)alloc((long)Dc * Dc * 2);
  _Float16* w_mv2 = (_Float16*)alloc((long)Vc * Dc * 2);
  _Float16* w_mq1 = (_Float16*)alloc((long)Dc * Dc * 2);
  _Float16* w_mq2 = (_Float16*)alloc((long)Kc * Dc * 2);
  _Float16* w_q   = (_Float16*)alloc((long)Dc * Dc * 2);
  _Float16* w_k   = (_Float16*)alloc((long)Dc * Dc * 2);
  _Float16* w_v   = (_Float16*)alloc((long)Dc * Dc * 2);
  _Float16* w_mr  = (_Float16*)alloc((long)Dc * Vc * 2);
  _Float16* w_out = (_Float16*)alloc((long)Dc * Dc * 2);

  auto tcvt = [&](const float* s, _Float16* d, int Kd_, int N_) {
    long n = (long)Kd_ * N_;
    transpose_f32_to_f16<<<(int)((n + 255) / 256), 256, 0, stream>>>(s, d, Kd_, N_);
  };
  auto gemm = [&](const _Float16* A, const _Float16* Wt, const float* bias,
                  const float* Cadd, float* o32, _Float16* o16,
                  int Mr, int Kd_, int N_, int act_) {
    dim3 g(Mr / 128, N_ / 128);
    gemm_wmma<<<g, 256, 0, stream>>>(A, Wt, bias, Cadd, o32, o16, Mr, Kd_, N_, act_);
  };

  // 0) weight convert + transpose (f32 KxN -> f16 NxK)
  tcvt(mk_w1, w_mk1, Dc, Dc);  tcvt(mk_w2, w_mk2, Dc, Kc);
  tcvt(mv_w1, w_mv1, Dc, Dc);  tcvt(mv_w2, w_mv2, Dc, Vc);
  tcvt(mq_w1, w_mq1, Dc, Dc);  tcvt(mq_w2, w_mq2, Dc, Kc);
  tcvt(q_w,   w_q,   Dc, Dc);  tcvt(k_w,   w_k,   Dc, Dc);
  tcvt(v_w,   w_v,   Dc, Dc);  tcvt(mr_w,  w_mr,  Vc, Dc);
  tcvt(out_w, w_out, Dc, Dc);

  // 1) layernorm
  ln_kernel<<<(int)BSc, 256, 0, stream>>>(x, ln_g, ln_b, xn16, Dc);

  // 2) write importance + per-batch top-k
  imp_kernel<<<(int)BSc, 256, 0, stream>>>(xn16, wg_w, wg_b, imp, Dc);
  topk_kernel<<<Bc, 1024, 0, stream>>>(imp, topidx, Sc, NWc);

  // 3) memory key/value MLPs (WMMA)
  gemm(xn16,  w_mk1, mk_b1, nullptr, nullptr, tmp16, (int)BSc, Dc, Dc, 1);
  gemm(tmp16, w_mk2, mk_b2, nullptr, mk32,   nullptr, (int)BSc, Dc, Kc, 0);
  gemm(xn16,  w_mv1, mv_b1, nullptr, nullptr, tmp16, (int)BSc, Dc, Dc, 1);
  gemm(tmp16, w_mv2, mv_b2, nullptr, mv32,   nullptr, (int)BSc, Dc, Vc, 0);

  // 4) updated memory + f16 stored keys / transposed values
  {
    long total = (long)Bc * Mc * KVc;
    build_mem_kernel<<<(int)((total + 255) / 256), 256, 0, stream>>>(
        mem, mk32, mv32, topidx, out_mem, sk16, svT16,
        Bc, Mc, KVc, Kc, KEEP, Sc, NWc);
  }

  // 5) memory queries + flash attention over memory (WMMA)
  gemm(xn16,  w_mq1, mq_b1, nullptr, nullptr, tmp16, (int)BSc, Dc, Dc, 1);
  gemm(tmp16, w_mq2, mq_b2, nullptr, nullptr, mq16,  (int)BSc, Dc, Kc, 0);
  {
    dim3 g(Sc / 64, Bc);
    flash_attn_kernel<<<g, 128, 0, stream>>>(mq16, sk16, svT16, ret16, Sc, Mc);
  }

  // 6) memory output projection (WMMA)
  gemm(ret16, w_mr, mr_b, nullptr, nullptr, mo16, (int)BSc, Vc, Dc, 0);

  // 7) q/k/v projections (WMMA)
  gemm(xn16, w_q, nullptr, nullptr, nullptr, q16,  (int)BSc, Dc, Dc, 0);
  gemm(xn16, w_k, nullptr, nullptr, nullptr, k16,  (int)BSc, Dc, Dc, 0);
  gemm(xn16, w_v, nullptr, nullptr, nullptr, v16b, (int)BSc, Dc, Dc, 0);

  // 8) global context: mean over S, then gp projection
  colmean_kernel<<<(Bc * Dc) / 256, 256, 0, stream>>>(q16, qmean, Sc, Dc);
  gp_kernel<<<(Bc * Dc) / 256, 256, 0, stream>>>(qmean, gp_w, gp_b, gctx, Dc);

  // 9) grouped conv + combine + position gate
  conv_combine_kernel<<<(int)BSc, 256, 0, stream>>>(
      v16b, conv_w, conv_b, gctx, mo16, k16, comb16, Sc, Dc, Dc / GR);

  // 10) output projection + residual (WMMA, f32 out)
  gemm(comb16, w_out, out_b, x, out_main, nullptr, (int)BSc, Dc, Dc, 0);
}
